// ScaledDotProductAttention_53687091200524
// MI455X (gfx1250) — matrix operand order repaired and checked by reading, not compiled, not run on hardware
//
#include <hip/hip_runtime.h>

typedef __attribute__((ext_vector_type(16))) _Float16 v16h;
typedef __attribute__((ext_vector_type(8)))  float    v8f;

#define B_ 4
#define H_ 16
#define S_ 1024
#define D_ 64
#define KSTR 66        // padded LDS row stride for K [s][d] (halfs) -> conflict-free B-frag reads
#define VSTR (S_ + 2)  // padded LDS row stride for V^T [d][s] (halfs)
#define NEG_FILL -1000000000.0f
#define INV_TEMP 0.125f

__global__ __launch_bounds__(256) void sdpa_wmma_kernel(
    const float* __restrict__ q, const float* __restrict__ k,
    const float* __restrict__ v, const float* __restrict__ sim,
    const unsigned char* __restrict__ mask,
    float* __restrict__ outp, float* __restrict__ attn)
{
    extern __shared__ char smem_raw[];
    _Float16* sK  = (_Float16*)smem_raw;  // phase 1: K  as [S_][KSTR] f16
    _Float16* sVt = (_Float16*)smem_raw;  // phase 3: V^T as [D_][VSTR] f16 (reused)

    const int tid  = threadIdx.x;
    const int wave = tid >> 5;
    const int lane = tid & 31;
    const int n    = lane & 15;   // tile column (B/C frags) and tile row (A frags)
    const int hi   = lane >> 4;   // half-wave select

    const int bh = blockIdx.x >> 3;     // b*H + h
    const int qb = blockIdx.x & 7;      // which 128-row q block
    const int b  = bh >> 4;             // H_ == 16

    const size_t bhSS = (size_t)bh * S_ * S_;
    const size_t bSS  = (size_t)b  * S_ * S_;
    const int qbase = qb * 128 + wave * 16;

    // ---------- cooperative load: K -> LDS (f32 -> f16), row-major [s][d] ----------
    {
        const float4* k4 = (const float4*)(k + (size_t)bh * S_ * D_);
        #pragma unroll 4
        for (int i = 0; i < (S_ * D_ / 4) / 256; ++i) {
            int fi = tid + i * 256;
            int s  = fi >> 4;       // D_/4 == 16 float4 per row
            int d4 = fi & 15;
            float4 kv = k4[fi];
            _Float16* dst = sK + s * KSTR + d4 * 4;
            dst[0] = (_Float16)kv.x; dst[1] = (_Float16)kv.y;
            dst[2] = (_Float16)kv.z; dst[3] = (_Float16)kv.w;
        }
    }
    __syncthreads();

    // ---------- load Q A-fragments (scaled by 1/temperature), f16, K-dim 0..31 / 32..63 ----------
    v16h aq0, aq1;
    {
        const float* qrow = q + ((size_t)bh * S_ + qbase + n) * D_;
        #pragma unroll
        for (int vv = 0; vv < 8; ++vv) {
            int kb = 2 * vv + 16 * hi;   // A 16x32 f16 layout
            aq0[2*vv]   = (_Float16)(qrow[kb]       * INV_TEMP);
            aq0[2*vv+1] = (_Float16)(qrow[kb + 1]   * INV_TEMP);
            aq1[2*vv]   = (_Float16)(qrow[32 + kb]     * INV_TEMP);
            aq1[2*vv+1] = (_Float16)(qrow[32 + kb + 1] * INV_TEMP);
        }
    }

    // ---------- phase 1: logits = QK^T (WMMA) + mask + similarity; write raw logits; track row max ----------
    float rmax[8];
    #pragma unroll
    for (int r = 0; r < 8; ++r) rmax[r] = -3.0e38f;

    for (int j = 0; j < S_ / 16; ++j) {
        v8f c = {};
        v16h bk;
        #pragma unroll
        for (int vv = 0; vv < 8; ++vv) {               // B 32x16: K grouped by 16 per half-wave
            const _Float16* src = sK + (j * 16 + n) * KSTR + 2 * vv + 16 * hi;
            bk[2*vv] = src[0]; bk[2*vv+1] = src[1];
        }
        c = __builtin_amdgcn_wmma_f32_16x16x32_f16(false, aq0, false, bk, (short)0, c, false, false);
        #pragma unroll
        for (int vv = 0; vv < 8; ++vv) {
            const _Float16* src = sK + (j * 16 + n) * KSTR + 32 + 2 * vv + 16 * hi;
            bk[2*vv] = src[0]; bk[2*vv+1] = src[1];
        }
        c = __builtin_amdgcn_wmma_f32_16x16x32_f16(false, aq1, false, bk, (short)0, c, false, false);

        const int col = j * 16 + n;
        #pragma unroll
        for (int r = 0; r < 8; ++r) {                  // C layout: M = r + 8*hi, N = n
            const int row = qbase + r + 8 * hi;
            float val = c[r];
            if (mask[bSS + (size_t)row * S_ + col]) val = NEG_FILL;
            val += sim[bhSS + (size_t)row * S_ + col];
            attn[bhSS + (size_t)row * S_ + col] = val;
            rmax[r] = fmaxf(rmax[r], val);
        }
    }
    #pragma unroll
    for (int off = 1; off < 16; off <<= 1)
        #pragma unroll
        for (int r = 0; r < 8; ++r)
            rmax[r] = fmaxf(rmax[r], __shfl_xor(rmax[r], off));

    // ---------- phase 2: softmax over L2-hot logits (each wave owns its 16 rows) ----------
    #pragma unroll 1
    for (int r = 0; r < 8; ++r) {
        const int row = qbase + r + 8 * hi;
        float* prow = attn + bhSS + (size_t)row * S_;
        const float m = rmax[r];
        float sum = 0.f;
        for (int c16 = 0; c16 < S_ / 16; ++c16) {
            int col = c16 * 16 + n;
            float e = __expf(prow[col] - m);
            prow[col] = e;
            sum += e;
        }
        #pragma unroll
        for (int off = 1; off < 16; off <<= 1) sum += __shfl_xor(sum, off);
        const float inv = 1.0f / sum;
        for (int c16 = 0; c16 < S_ / 16; ++c16) {
            int col = c16 * 16 + n;
            prow[col] *= inv;
        }
    }

    // ---------- swap LDS contents: V^T -> LDS [d][s] f16 ----------
    __syncthreads();   // everyone done reading sK
    {
        const float4* v4 = (const float4*)(v + (size_t)bh * S_ * D_);
        #pragma unroll 4
        for (int i = 0; i < (S_ * D_ / 4) / 256; ++i) {
            int fi = tid + i * 256;
            int s  = fi >> 4;
            int d4 = fi & 15;
            float4 vv = v4[fi];
            sVt[(d4 * 4 + 0) * VSTR + s] = (_Float16)vv.x;
            sVt[(d4 * 4 + 1) * VSTR + s] = (_Float16)vv.y;
            sVt[(d4 * 4 + 2) * VSTR + s] = (_Float16)vv.z;
            sVt[(d4 * 4 + 3) * VSTR + s] = (_Float16)vv.w;
        }
    }
    __syncthreads();

    // ---------- phase 3: output = P @ V (WMMA), P re-read from L2-hot attn ----------
    v8f acc[4] = {{}, {}, {}, {}};
    const float* prowA = attn + bhSS + (size_t)(qbase + n) * S_;
    for (int kc = 0; kc < S_ / 32; ++kc) {
        v16h pa;
        #pragma unroll
        for (int vv = 0; vv < 8; ++vv) {
            int kb = 2 * vv + 16 * hi;
            const float* pp = prowA + kc * 32 + kb;
            pa[2*vv]   = (_Float16)pp[0];
            pa[2*vv+1] = (_Float16)pp[1];
        }
        #pragma unroll
        for (int j = 0; j < 4; ++j) {                  // D_ = 64 -> four 16-col tiles
            v16h bv;
            #pragma unroll
            for (int vv = 0; vv < 8; ++vv) {
                const _Float16* src = sVt + (j * 16 + n) * VSTR + kc * 32 + 2 * vv + 16 * hi;
                bv[2*vv] = src[0]; bv[2*vv+1] = src[1];
            }
            acc[j] = __builtin_amdgcn_wmma_f32_16x16x32_f16(false, pa, false, bv, (short)0, acc[j], false, false);
        }
    }

    #pragma unroll
    for (int j = 0; j < 4; ++j)
        #pragma unroll
        for (int r = 0; r < 8; ++r) {
            const int row = qbase + r + 8 * hi;
            outp[((size_t)bh * S_ + row) * D_ + j * 16 + n] = acc[j][r];
        }
}

extern "C" void kernel_launch(void* const* d_in, const int* in_sizes, int n_in,
                              void* d_out, int out_size, void* d_ws, size_t ws_size,
                              hipStream_t stream) {
    const float*         q    = (const float*)d_in[0];
    const float*         k    = (const float*)d_in[1];
    const float*         v    = (const float*)d_in[2];
    const float*         sim  = (const float*)d_in[3];
    const unsigned char* mask = (const unsigned char*)d_in[4];  // jnp.bool_: 1 byte

    float* outp = (float*)d_out;                                // [B,H,S,D]
    float* attn = outp + (size_t)B_ * H_ * S_ * D_;             // [B,H,S,S]

    const size_t shmem = (size_t)S_ * KSTR * sizeof(_Float16);  // 132 KB (>= D_*VSTR*2)
    dim3 grid(B_ * H_ * (S_ / 128));                            // 512 workgroups
    sdpa_wmma_kernel<<<grid, 256, shmem, stream>>>(q, k, v, sim, mask, outp, attn);
}
